// Step3p5MoEMLP_82145544503834
// MI455X (gfx1250) — matrix operand order repaired and checked by reading, not compiled, not run on hardware
//
#include <hip/hip_runtime.h>
#include <hip/hip_bf16.h>

#define T_DIM 2048
#define H_DIM 2048
#define E_DIM 16
#define I_DIM 512
#define K_TOP 2
#define ROUTED_SCALE 1.5f
#define SWIGLU_LIMIT 7.0f
#define TILE_M 32
#define MAX_TILES 64    // T / TILE_M (worst case: all tokens to one expert)

typedef __attribute__((ext_vector_type(16))) __bf16 v16bf;
typedef __attribute__((ext_vector_type(8)))  __bf16 v8bf;
typedef __attribute__((ext_vector_type(8)))  float  v8f;

// ---------- bf16 helpers (round-to-nearest-even) ----------
__device__ __forceinline__ __bf16 f2bf(float f) {
    unsigned u = __builtin_bit_cast(unsigned, f);
    unsigned r = u + 0x7fffu + ((u >> 16) & 1u);
    unsigned short h = (unsigned short)(r >> 16);
    return __builtin_bit_cast(__bf16, h);
}

// swizzle map for A-operand chunk order within a 32-element K block:
// stored order {K0-7, K8-15, K16-23, K24-31}  (identity)
__device__ __forceinline__ int chunk_map(int c) {
    // {0,1,2,3}
    return c;
}

// ---------- kernel 0: zero per-expert counters ----------
__global__ void zero_cnt_kernel(int* __restrict__ cnt) {
    if (threadIdx.x < E_DIM) cnt[threadIdx.x] = 0;
}

// ---------- kernel 1: fp32 -> bf16 hidden states, A-swizzled ----------
// one thread per 8-element chunk: out chunk c takes source chunk chunk_map(c&3)
__global__ void convert_x_kernel(const float* __restrict__ hs, __bf16* __restrict__ xb) {
    int c = blockIdx.x * 256 + threadIdx.x;          // chunk index, T*H/8 total
    if (c >= (T_DIM * H_DIM) / 8) return;
    long srcBase = (long)(c >> 2) * 32 + chunk_map(c & 3) * 8;
    float4 f0 = *(const float4*)(hs + srcBase);
    float4 f1 = *(const float4*)(hs + srcBase + 4);
    v8bf o;
    o[0] = f2bf(f0.x); o[1] = f2bf(f0.y); o[2] = f2bf(f0.z); o[3] = f2bf(f0.w);
    o[4] = f2bf(f1.x); o[5] = f2bf(f1.y); o[6] = f2bf(f1.z); o[7] = f2bf(f1.w);
    *(v8bf*)(xb + (long)c * 8) = o;
}

// ---------- kernel 1b: plain fp32 -> bf16 weight conversion ----------
__global__ void convert_w_kernel(const float* __restrict__ w, __bf16* __restrict__ wb, long nchunks) {
    long c = (long)blockIdx.x * 256 + threadIdx.x;   // 8 elements per thread
    if (c >= nchunks) return;
    float4 f0 = *(const float4*)(w + c * 8);
    float4 f1 = *(const float4*)(w + c * 8 + 4);
    v8bf o;
    o[0] = f2bf(f0.x); o[1] = f2bf(f0.y); o[2] = f2bf(f0.z); o[3] = f2bf(f0.w);
    o[4] = f2bf(f1.x); o[5] = f2bf(f1.y); o[6] = f2bf(f1.z); o[7] = f2bf(f1.w);
    *(v8bf*)(wb + c * 8) = o;
}

// ---------- kernel 2: router (one wave per token) ----------
__global__ __launch_bounds__(32) void router_kernel(
    const float* __restrict__ hs, const float* __restrict__ gw,
    const float* __restrict__ bias, int* __restrict__ cnt,
    int* __restrict__ listTok, int* __restrict__ listSlot,
    float* __restrict__ listW)
{
    int t = blockIdx.x;
    int lane = threadIdx.x;

    float part[E_DIM];
#pragma unroll
    for (int e = 0; e < E_DIM; ++e) part[e] = 0.f;

    const float* xrow = hs + (long)t * H_DIM;
    for (int h = lane; h < H_DIM; h += 32) {
        float x = xrow[h];
#pragma unroll
        for (int e = 0; e < E_DIM; ++e) part[e] += x * gw[e * H_DIM + h];
    }
#pragma unroll
    for (int off = 16; off > 0; off >>= 1) {
#pragma unroll
        for (int e = 0; e < E_DIM; ++e) part[e] += __shfl_xor(part[e], off, 32);
    }

    if (lane == 0) {
        float sc[E_DIM], bsc[E_DIM];
#pragma unroll
        for (int e = 0; e < E_DIM; ++e) {
            sc[e] = 1.f / (1.f + __expf(-part[e]));
            bsc[e] = sc[e] + bias[e];
        }
        int e0 = 0;
#pragma unroll
        for (int e = 1; e < E_DIM; ++e) if (bsc[e] > bsc[e0]) e0 = e;
        int e1 = (e0 == 0) ? 1 : 0;
#pragma unroll
        for (int e = 0; e < E_DIM; ++e) if (e != e0 && bsc[e] > bsc[e1]) e1 = e;

        float s0 = sc[e0], s1 = sc[e1];
        float inv = ROUTED_SCALE / (s0 + s1);

        int p0 = atomicAdd(&cnt[e0], 1);
        listTok[e0 * T_DIM + p0]  = t;
        listSlot[e0 * T_DIM + p0] = t * K_TOP + 0;
        listW[e0 * T_DIM + p0]    = s0 * inv;

        int p1 = atomicAdd(&cnt[e1], 1);
        listTok[e1 * T_DIM + p1]  = t;
        listSlot[e1 * T_DIM + p1] = t * K_TOP + 1;
        listW[e1 * T_DIM + p1]    = s1 * inv;
    }
}

// ---------- kernel 3: expert MLP, all-bf16 WMMA, TILE_M=32 ----------
__global__ __launch_bounds__(256) void moe_expert_kernel(
    const __bf16* __restrict__ w13b, const __bf16* __restrict__ w2b,
    const __bf16* __restrict__ xb, const int* __restrict__ cnt,
    const int* __restrict__ listTok, const int* __restrict__ listSlot,
    const float* __restrict__ listW, float* __restrict__ pairout)
{
    __shared__ __bf16 s_act[TILE_M][I_DIM];   // 32 KB, A-swizzled layout
    __shared__ int    s_tok[TILE_M];
    __shared__ int    s_slot[TILE_M];
    __shared__ float  s_w[TILE_M];
    __shared__ int    s_valid[TILE_M];

    int e = blockIdx.x;
    int tile = blockIdx.y;
    int n_tok_e = cnt[e];
    if (tile * TILE_M >= n_tok_e) return;

    int tid = threadIdx.x;
    if (tid < TILE_M) {
        int idx = tile * TILE_M + tid;
        bool v = idx < n_tok_e;
        s_tok[tid]   = v ? listTok[e * T_DIM + idx]  : 0;
        s_slot[tid]  = v ? listSlot[e * T_DIM + idx] : 0;
        s_w[tid]     = v ? listW[e * T_DIM + idx]    : 0.f;
        s_valid[tid] = v ? 1 : 0;
    }
    __syncthreads();

    int wave = tid >> 5;       // 0..7
    int lane = tid & 31;
    int nlo  = lane & 15;      // N column / A row within 16
    int hi   = lane >> 4;      // K-half selector

    const __bf16* arow0 = xb + (long)s_tok[nlo] * H_DIM;        // M-subtile 0
    const __bf16* arow1 = xb + (long)s_tok[16 + nlo] * H_DIM;   // M-subtile 1

    // ---- GEMM1 + SwiGLU: act[32 x 512], g and u columns computed together ----
    for (int nt = wave; nt < I_DIM / 16; nt += 8) {
        int ncol = nt * 16 + nlo;
        const __bf16* wg = w13b + ((long)e * (2 * I_DIM) + ncol) * H_DIM;
        const __bf16* wu = w13b + ((long)e * (2 * I_DIM) + I_DIM + ncol) * H_DIM;
        v8f accg0 = {}, accg1 = {}, accu0 = {}, accu1 = {};
        for (int k0 = 0; k0 < H_DIM; k0 += 32) {
            int ko = k0 + hi * 16;
            v16bf A0 = *(const v16bf*)(arow0 + ko);
            v16bf A1 = *(const v16bf*)(arow1 + ko);
            v16bf Bg = *(const v16bf*)(wg + ko);
            v16bf Bu = *(const v16bf*)(wu + ko);
            accg0 = __builtin_amdgcn_wmma_f32_16x16x32_bf16(false, A0, false, Bg, (short)0, accg0, false, false);
            accg1 = __builtin_amdgcn_wmma_f32_16x16x32_bf16(false, A1, false, Bg, (short)0, accg1, false, false);
            accu0 = __builtin_amdgcn_wmma_f32_16x16x32_bf16(false, A0, false, Bu, (short)0, accu0, false, false);
            accu1 = __builtin_amdgcn_wmma_f32_16x16x32_bf16(false, A1, false, Bu, (short)0, accu1, false, false);
        }
        // swizzled column for the A-operand layout of GEMM2
        int blk  = nt >> 1;
        int coff = (nt & 1) * 16 + nlo;
        int scol = blk * 32 + chunk_map(coff >> 3) * 8 + (coff & 7);
#pragma unroll
        for (int r = 0; r < 8; ++r) {
            int m = r + hi * 8;
            {
                float g = fminf(accg0[r], SWIGLU_LIMIT);
                float u = fminf(fmaxf(accu0[r], -SWIGLU_LIMIT), SWIGLU_LIMIT);
                s_act[m][scol] = f2bf(g * (1.f / (1.f + __expf(-g))) * u);
            }
            {
                float g = fminf(accg1[r], SWIGLU_LIMIT);
                float u = fminf(fmaxf(accu1[r], -SWIGLU_LIMIT), SWIGLU_LIMIT);
                s_act[16 + m][scol] = f2bf(g * (1.f / (1.f + __expf(-g))) * u);
            }
        }
    }
    __syncthreads();

    // ---- GEMM2: out[32 x H] = act[32 x I] * W2^T, scaled into pairout ----
    const __bf16* act0 = &s_act[nlo][0];
    const __bf16* act1 = &s_act[16 + nlo][0];
    for (int nt = wave; nt < H_DIM / 16; nt += 8) {
        int ncol = nt * 16 + nlo;
        const __bf16* wrow = w2b + ((long)e * H_DIM + ncol) * I_DIM;
        v8f acc0 = {}, acc1 = {};
        for (int k0 = 0; k0 < I_DIM; k0 += 32) {
            int ko = k0 + hi * 16;
            v16bf A0 = *(const v16bf*)(act0 + ko);
            v16bf A1 = *(const v16bf*)(act1 + ko);
            v16bf B  = *(const v16bf*)(wrow + ko);
            acc0 = __builtin_amdgcn_wmma_f32_16x16x32_bf16(false, A0, false, B, (short)0, acc0, false, false);
            acc1 = __builtin_amdgcn_wmma_f32_16x16x32_bf16(false, A1, false, B, (short)0, acc1, false, false);
        }
#pragma unroll
        for (int r = 0; r < 8; ++r) {
            int m = r + hi * 8;
            if (s_valid[m])
                pairout[(long)s_slot[m] * H_DIM + ncol] = s_w[m] * acc0[r];
            if (s_valid[16 + m])
                pairout[(long)s_slot[16 + m] * H_DIM + ncol] = s_w[16 + m] * acc1[r];
        }
    }
}

// ---------- kernel 4: combine the K=2 pair outputs ----------
__global__ void combine_kernel(const float* __restrict__ pairout, float* __restrict__ out) {
    int i = blockIdx.x * 256 + threadIdx.x;
    if (i < T_DIM * H_DIM) {
        int t = i / H_DIM;
        int h = i - t * H_DIM;
        out[i] = pairout[(long)(2 * t) * H_DIM + h] +
                 pairout[(long)(2 * t + 1) * H_DIM + h];
    }
}

extern "C" void kernel_launch(void* const* d_in, const int* in_sizes, int n_in,
                              void* d_out, int out_size, void* d_ws, size_t ws_size,
                              hipStream_t stream) {
    const float* hs   = (const float*)d_in[0];   // [T, H]
    const float* gw   = (const float*)d_in[1];   // [E, H]
    const float* bias = (const float*)d_in[2];   // [E]
    const float* w13  = (const float*)d_in[3];   // [E, 2I, H]
    const float* w2   = (const float*)d_in[4];   // [E, H, I]
    float* out = (float*)d_out;

    char* ws = (char*)d_ws;
    size_t off = 0;
    auto alloc = [&](size_t bytes) {
        void* p = ws + off;
        off = (off + bytes + 255) & ~(size_t)255;
        return p;
    };
    __bf16* xb      = (__bf16*)alloc((size_t)T_DIM * H_DIM * 2);                    // 8 MB
    __bf16* w13b    = (__bf16*)alloc((size_t)E_DIM * 2 * I_DIM * H_DIM * 2);        // 67 MB
    __bf16* w2b     = (__bf16*)alloc((size_t)E_DIM * H_DIM * I_DIM * 2);            // 33.5 MB
    float*  pairout = (float*) alloc((size_t)T_DIM * K_TOP * H_DIM * 4);            // 33.5 MB
    int*    cnt     = (int*)   alloc(E_DIM * 4);
    int*    listTok = (int*)   alloc((size_t)E_DIM * T_DIM * 4);
    int*    listSlot= (int*)   alloc((size_t)E_DIM * T_DIM * 4);
    float*  listW   = (float*) alloc((size_t)E_DIM * T_DIM * 4);

    long nc13 = (long)E_DIM * 2 * I_DIM * H_DIM / 8;
    long nc2  = (long)E_DIM * H_DIM * I_DIM / 8;

    zero_cnt_kernel<<<1, 32, 0, stream>>>(cnt);
    convert_x_kernel<<<(T_DIM * H_DIM / 8 + 255) / 256, 256, 0, stream>>>(hs, xb);
    convert_w_kernel<<<(int)((nc13 + 255) / 256), 256, 0, stream>>>(w13, w13b, nc13);
    convert_w_kernel<<<(int)((nc2  + 255) / 256), 256, 0, stream>>>(w2,  w2b,  nc2);
    router_kernel<<<T_DIM, 32, 0, stream>>>(hs, gw, bias, cnt, listTok, listSlot, listW);
    moe_expert_kernel<<<dim3(E_DIM, MAX_TILES), 256, 0, stream>>>(
        w13b, w2b, xb, cnt, listTok, listSlot, listW, pairout);
    combine_kernel<<<(T_DIM * H_DIM + 255) / 256, 256, 0, stream>>>(pairout, out);
}
